// Res_MCNN_branch2_85358180041417
// MI455X (gfx1250) — matrix operand order repaired and checked by reading, not compiled, not run on hardware
//
#include <hip/hip_runtime.h>

typedef _Float16 f16;
typedef __attribute__((ext_vector_type(4)))  _Float16 v4h;
typedef __attribute__((ext_vector_type(16))) _Float16 v16h;
typedef __attribute__((ext_vector_type(8)))  float    v8f;

// LDS offset (relative to row 0 / lane pixel base) for im2col index k:
//   k = (ky*KW + kx)*CIN + c  ->  ky*SSTR + kx*CIN + c
// Called with compile-time k after full unrolling, so it folds to a constant.
template<int CIN, int KW, int SSTR>
__device__ __forceinline__ constexpr int loff(int k) {
  const int tap = k / CIN;
  const int c   = k - tap * CIN;
  const int ky  = tap / KW;
  const int kx  = tap - ky * KW;
  return ky * SSTR + kx * CIN + c;
}

// ---------------------------------------------------------------------------
// Implicit-GEMM conv (+ReLU, optionally fused 2x2 maxpool) via WMMA f16.
// Input patch staged in LDS (zero-padded halo, uint2 fast path when interior);
// B pre-packed in WMMA fragment order (one 32B vector load per lane).
// A fragments read from LDS as 4 aligned 8-byte vector loads per lane:
// CIN % 4 == 0 guarantees each 4-consecutive-K group (k ≡ 0 mod 4) stays in
// one tap and is 8B aligned. K tiles fully unrolled -> constant LDS offsets.
// ---------------------------------------------------------------------------
template<int CIN, int KH, int KW, int COUT, int NT, bool POOL, int H, int W>
__global__ __launch_bounds__(256) void conv_wmma(
    const f16*  __restrict__ in,
    const f16*  __restrict__ Bfrag,   // [KT][NT][32 lanes][16 halves]
    const float* __restrict__ bias,
    f16* __restrict__ out,
    int NB)
{
  static_assert(CIN % 4 == 0, "v4h A-gather needs CIN % 4 == 0");
  constexpr int KTOT  = KH * KW * CIN;      // multiple of 4
  constexpr int KT    = (KTOT + 31) / 32;
  constexpr int PH = KH / 2, PW = KW / 2;
  constexpr int ROWS  = POOL ? 2 : 1;
  constexpr int SROWS = ROWS + KH - 1;      // staged input rows
  constexpr int SW    = 128 + KW - 1;       // staged input columns
  constexpr int SSTR  = SW * CIN;           // halves per staged row (mult of 4)
  __shared__ __align__(16) f16 smem[SROWS * SSTR];

  const int tid  = threadIdx.x;
  const int lane = tid & 31;
  const int wv   = tid >> 5;

  constexpr int XSEG = W / 128;
  constexpr int HT   = POOL ? (H / 2) : H;
  const int xseg = blockIdx.x % XSEG;       // power-of-2 -> mask
  const int rest = blockIdx.x / XSEG;
  const int yt = rest % HT;
  const int b  = rest / HT;
  const int y0  = POOL ? (2 * yt) : yt;
  const int gx0 = xseg * 128 - PW;          // global x of LDS column 0
  const int gy0 = y0 - PH;                  // global y of LDS row 0

  // ---- cooperative staging: global NHWC -> LDS, halo zero-filled ---------
  const bool interior = (gy0 >= 0) & (gy0 + SROWS <= H) &
                        (gx0 >= 0) & (gx0 + SW <= W);
  if (interior) {                           // block-uniform branch
    constexpr int QROW = SSTR / 4;          // uint2 chunks per row
    for (int e = tid; e < SROWS * QROW; e += 256) {
      const int row = e / QROW;
      const int q   = e - row * QROW;
      const size_t gi = ((size_t)(b * H + gy0 + row) * W + gx0) * CIN + q * 4;
      *(uint2*)(smem + row * SSTR + q * 4) = *(const uint2*)(in + gi);
    }
  } else {
    for (int e = tid; e < SROWS * SSTR; e += 256) {
      const int row = e / SSTR;
      const int xc  = e - row * SSTR;
      const int xq  = xc / CIN;
      const int cc  = xc - xq * CIN;
      const int yi  = gy0 + row;
      const int xi  = gx0 + xq;
      f16 v = (f16)0.f;
      if (yi >= 0 && yi < H && xi >= 0 && xi < W)
        v = in[((size_t)(b * H + yi) * W + xi) * CIN + cc];
      smem[e] = v;
    }
  }
  __syncthreads();

  v8f zero = {};
  v8f acc[NT][ROWS];
#pragma unroll
  for (int nt = 0; nt < NT; ++nt)
#pragma unroll
    for (int r = 0; r < ROWS; ++r)
      acc[nt][r] = zero;

  // ISA 16-bit A layout (16x32): m = lane&15; half h ->
  // K = (h&7) + (h>=8 ? 16 : 0) + (lane>=16 ? 8 : 0)
  // Grouped: halves [4g..4g+3] <-> K base {0,4,16,20}[g] (+8 upper half-wave).
  const int m    = lane & 15;
  const int lxC  = (wv * 16 + m) * CIN;     // per-lane pixel base (mult of 4)
  const bool hi  = (lane & 16) != 0;
  const v16h* __restrict__ Bv = (const v16h*)Bfrag;
  const v4h zero4 = {};

  union AFrag { v16h v; v4h q[4]; };

#pragma unroll
  for (int kt = 0; kt < KT; ++kt) {
    if (kt + 1 < KT)  // hint next B block into cache (global_prefetch_b8)
      __builtin_prefetch((const char*)(Bv + (size_t)(kt + 1) * NT * 32), 0, 1);

    v16h a[ROWS];
#pragma unroll
    for (int r = 0; r < ROWS; ++r) {
      AFrag af;
#pragma unroll
      for (int g = 0; g < 4; ++g) {
        const int kbase = kt * 32 + ((g & 1) ? 4 : 0) + ((g & 2) ? 16 : 0);
        const int kA = kbase;      // lanes 0-15
        const int kB = kbase + 8;  // lanes 16-31
        v4h q = zero4;
        if (kA < KTOT) {                     // compile-time condition
          const int offA = loff<CIN, KW, SSTR>(kA) + r * SSTR;
          const int offB = (kB < KTOT) ? (loff<CIN, KW, SSTR>(kB) + r * SSTR)
                                       : offA;
          const int off  = hi ? offB : offA; // single cndmask
          const v4h lv = *(const v4h*)(smem + lxC + off);
          if (kB < KTOT) q = lv;             // compile-time
          else           q = hi ? zero4 : lv;
        }
        af.q[g] = q;
      }
      a[r] = af.v;
    }

#pragma unroll
    for (int nt = 0; nt < NT; ++nt) {
      const v16h bf = Bv[(size_t)(kt * NT + nt) * 32 + lane];
#pragma unroll
      for (int r = 0; r < ROWS; ++r)
        acc[nt][r] = __builtin_amdgcn_wmma_f32_16x16x32_f16(
            false, a[r], false, bf, (short)0, acc[nt][r], false, false);
    }
  }

  // D layout: vgpr r holds pixel M = r (+8 for lanes>=16), N(cout) = lane&15.
  const int nl = lane & 15;
  const int xt = xseg * 8 + wv;             // this wave's M-tile index
  if constexpr (POOL) {
    constexpr int Ho = H / 2, Wo = W / 2;
    const int xo = xt * 8 + (hi ? 4 : 0);
#pragma unroll
    for (int nt = 0; nt < NT; ++nt) {
      const int n = nt * 16 + nl;
      if (n < COUT) {
        const float bv = bias[n];
#pragma unroll
        for (int p = 0; p < 4; ++p) {
          float z = fmaxf(fmaxf(acc[nt][0][2 * p], acc[nt][0][2 * p + 1]),
                          fmaxf(acc[nt][1][2 * p], acc[nt][1][2 * p + 1]));
          float vv = fmaxf(z + bv, 0.f);
          out[((size_t)(b * Ho + yt) * Wo + (xo + p)) * COUT + n] = (f16)vv;
        }
      }
    }
  } else {
#pragma unroll
    for (int nt = 0; nt < NT; ++nt) {
      const int n = nt * 16 + nl;
      if (n < COUT) {
        const float bv = bias[n];
#pragma unroll
        for (int r = 0; r < 8; ++r) {
          const int mm = (hi ? 8 : 0) + r;
          float vv = fmaxf(acc[nt][0][r] + bv, 0.f);
          out[((size_t)(b * H + y0) * W + (xt * 16 + mm)) * COUT + n] = (f16)vv;
        }
      }
    }
  }
}

// ---------------------------------------------------------------------------
// Convert f32 NCHW input -> f16 NHWC (4 channels, channel 3 zero-padded).
// ---------------------------------------------------------------------------
__global__ void pack_input(const float* __restrict__ in, f16* __restrict__ out,
                           int NB, int H, int W) {
  const size_t total = (size_t)NB * H * W;
  const size_t i = (size_t)blockIdx.x * blockDim.x + threadIdx.x;
  if (i >= total) return;
  const size_t HW  = (size_t)H * W;
  const size_t bb  = i / HW;
  const size_t rem = i - bb * HW;
  const float* base = in + bb * 3 * HW + rem;
  out[i * 4 + 0] = (f16)base[0];
  out[i * 4 + 1] = (f16)base[HW];
  out[i * 4 + 2] = (f16)base[2 * HW];
  out[i * 4 + 3] = (f16)0.f;
}

// ---------------------------------------------------------------------------
// Pack OIHW f32 weights -> f16 B operand in WMMA fragment order:
//   index = (((kt*NT + nt)*32 + lane)*16 + h)
//   n = nt*16 + (lane&15),  K = kt*32 + (h&7) + (h&8 ? 16 : 0) + (lane&16 ? 8 : 0)
//   K -> tap = K/CIN_EN, c = K%CIN_EN ; zero outside valid ranges.
// ---------------------------------------------------------------------------
__global__ void pack_weights_frag(const float* __restrict__ w, f16* __restrict__ Bf,
                                  int COUT, int CIN_REAL, int CIN_EN,
                                  int KH, int KW, int KT, int NT) {
  const int total = KT * NT * 32 * 16;
  const int i = blockIdx.x * blockDim.x + threadIdx.x;
  if (i >= total) return;
  const int h    = i & 15;
  const int lane = (i >> 4) & 31;
  const int blk  = i >> 9;
  const int nt   = blk % NT;
  const int kt   = blk / NT;

  const int n = nt * 16 + (lane & 15);
  const int K = kt * 32 + (h & 7) + ((h & 8) ? 16 : 0) + ((lane & 16) ? 8 : 0);
  const int KTOT = KH * KW * CIN_EN;

  f16 v = (f16)0.f;
  if (K < KTOT && n < COUT) {
    const int tap = K / CIN_EN;
    const int c   = K - tap * CIN_EN;
    if (c < CIN_REAL) {
      const int ky = tap / KW;
      const int kx = tap - ky * KW;
      v = (f16)w[(((size_t)n * CIN_REAL + c) * KH + ky) * KW + kx];
    }
  }
  Bf[i] = v;
}

// ---------------------------------------------------------------------------
// conv5 (1x1, 10->1) + gray residual + final relu.
// 4x bilinear downscale with half-pixel centers == mean of the 2x2 block at
// rows {4i+1,4i+2}, cols {4j+1,4j+2} (all bilinear weights are exactly 0.5).
// ---------------------------------------------------------------------------
__global__ void final_fuse(const f16* __restrict__ a4,
                           const float* __restrict__ w5,
                           const float* __restrict__ b5,
                           const float* __restrict__ input,
                           float* __restrict__ out, int NB) {
  const int Ho = 192, Wo = 256, H = 768, W = 1024;
  const size_t total = (size_t)NB * Ho * Wo;
  const size_t i = (size_t)blockIdx.x * blockDim.x + threadIdx.x;
  if (i >= total) return;
  const int j = (int)(i % Wo);
  const int t = (int)(i / Wo);
  const int y = t % Ho;
  const int b = t / Ho;

  float s = b5[0];
#pragma unroll
  for (int c = 0; c < 10; ++c)
    s += (float)a4[i * 10 + c] * w5[c];

  const size_t HW = (size_t)H * W;
  const float* R  = input + (size_t)b * 3 * HW;
  const float* G  = R + HW;
  const float* Bc = R + 2 * HW;
  const int yy = 4 * y + 1, xx = 4 * j + 1;
  float g = 0.f;
#pragma unroll
  for (int dy = 0; dy < 2; ++dy)
#pragma unroll
    for (int dx = 0; dx < 2; ++dx) {
      const size_t idx = (size_t)(yy + dy) * W + (xx + dx);
      g += 0.299f * R[idx] + 0.587f * G[idx] + 0.114f * Bc[idx];
    }
  g *= 0.25f;
  out[i] = fmaxf(s + fmaxf(g, 0.f), 0.f);
}

// ---------------------------------------------------------------------------
extern "C" void kernel_launch(void* const* d_in, const int* in_sizes, int n_in,
                              void* d_out, int out_size, void* d_ws, size_t ws_size,
                              hipStream_t stream) {
  const float* input = (const float*)d_in[0];
  const float* w1 = (const float*)d_in[1];
  const float* b1 = (const float*)d_in[2];
  const float* w2 = (const float*)d_in[3];
  const float* b2 = (const float*)d_in[4];
  const float* w3 = (const float*)d_in[5];
  const float* b3 = (const float*)d_in[6];
  const float* w4 = (const float*)d_in[7];
  const float* b4 = (const float*)d_in[8];
  const float* w5 = (const float*)d_in[9];
  const float* b5 = (const float*)d_in[10];

  const int H = 768, W = 1024;
  const int NB = in_sizes[0] / (3 * H * W);   // batch = 8

  // workspace carve-out (all region sizes are multiples of 256 B)
  char* ws = (char*)d_ws;
  size_t off = 0;
  f16* x0 = (f16*)(ws + off); off += (size_t)NB * H * W * 4 * 2;            // 8,768,1024,4
  f16* p1 = (f16*)(ws + off); off += (size_t)NB * (H/2) * (W/2) * 20 * 2;   // 8,384,512,20
  f16* p2 = (f16*)(ws + off); off += (size_t)NB * (H/4) * (W/4) * 40 * 2;   // 8,192,256,40
  f16* a3 = (f16*)(ws + off); off += (size_t)NB * (H/4) * (W/4) * 20 * 2;   // 8,192,256,20
  f16* a4 = (f16*)(ws + off); off += (size_t)NB * (H/4) * (W/4) * 10 * 2;   // 8,192,256,10
  f16* Bm1 = (f16*)(ws + off); off += (size_t)7  * 2 * 512 * 2;  // KT=7,  NT=2
  f16* Bm2 = (f16*)(ws + off); off += (size_t)16 * 3 * 512 * 2;  // KT=16, NT=3
  f16* Bm3 = (f16*)(ws + off); off += (size_t)32 * 2 * 512 * 2;  // KT=32, NT=2
  f16* Bm4 = (f16*)(ws + off); off += (size_t)16 * 1 * 512 * 2;  // KT=16, NT=1

  // ---- prep: input layout + weight packing -------------------------------
  {
    const size_t total = (size_t)NB * H * W;
    pack_input<<<(unsigned)((total + 255) / 256), 256, 0, stream>>>(input, x0, NB, H, W);
  }
  pack_weights_frag<<<(7  * 2 * 512 + 255) / 256, 256, 0, stream>>>(w1, Bm1, 20,  3,  4, 7, 7, 7,  2);
  pack_weights_frag<<<(16 * 3 * 512 + 255) / 256, 256, 0, stream>>>(w2, Bm2, 40, 20, 20, 5, 5, 16, 3);
  pack_weights_frag<<<(32 * 2 * 512 + 255) / 256, 256, 0, stream>>>(w3, Bm3, 20, 40, 40, 5, 5, 32, 2);
  pack_weights_frag<<<(16 * 1 * 512 + 255) / 256, 256, 0, stream>>>(w4, Bm4, 10, 20, 20, 5, 5, 16, 1);

  // ---- conv1 (3->20, 7x7) + relu + 2x2 maxpool, WMMA ---------------------
  {
    const int blocks = NB * (768 / 2) * (1024 / 128);       // 24576
    conv_wmma<4, 7, 7, 20, 2, true, 768, 1024><<<blocks, 256, 0, stream>>>(x0, Bm1, b1, p1, NB);
  }
  // ---- conv2 (20->40, 5x5) + relu + 2x2 maxpool, WMMA --------------------
  {
    const int blocks = NB * (384 / 2) * (512 / 128);        // 6144
    conv_wmma<20, 5, 5, 40, 3, true, 384, 512><<<blocks, 256, 0, stream>>>(p1, Bm2, b2, p2, NB);
  }
  // ---- conv3 (40->20, 5x5) + relu, WMMA ----------------------------------
  {
    const int blocks = NB * 192 * (256 / 128);              // 3072
    conv_wmma<40, 5, 5, 20, 2, false, 192, 256><<<blocks, 256, 0, stream>>>(p2, Bm3, b3, a3, NB);
  }
  // ---- conv4 (20->10, 5x5) + relu, WMMA ----------------------------------
  {
    const int blocks = NB * 192 * (256 / 128);              // 3072
    conv_wmma<20, 5, 5, 10, 1, false, 192, 256><<<blocks, 256, 0, stream>>>(a3, Bm4, b4, a4, NB);
  }
  // ---- conv5 (1x1) + gray residual + relu --------------------------------
  {
    const size_t total = (size_t)NB * 192 * 256;            // 393216
    final_fuse<<<(unsigned)((total + 255) / 256), 256, 0, stream>>>(a4, w5, b5, input, (float*)d_out, NB);
  }
  (void)n_in; (void)out_size; (void)ws_size;
}
